// AttentionWithKVCache_11458972745861
// MI455X (gfx1250) — compile-verified, matrix operand order checked by reading, not run on hardware
//
#include <hip/hip_runtime.h>

typedef __attribute__((ext_vector_type(16))) _Float16 v16h;
typedef __attribute__((ext_vector_type(8)))  _Float16 v8h;
typedef __attribute__((ext_vector_type(4)))  _Float16 v4h;
typedef __attribute__((ext_vector_type(8)))  float    v8f;
typedef __attribute__((ext_vector_type(4)))  int      v4i;

#ifndef __has_builtin
#define __has_builtin(x) 0
#endif

#define GAS __attribute__((address_space(1)))
#define LAS __attribute__((address_space(3)))

#if __has_builtin(__builtin_amdgcn_global_load_async_to_lds_b128)
#define HAVE_ASYNC_LDS 1
#else
#define HAVE_ASYNC_LDS 0
#endif

__device__ inline void async_cp16(const _Float16* g, _Float16* l) {
#if HAVE_ASYNC_LDS
    __builtin_amdgcn_global_load_async_to_lds_b128((GAS v4i*)g, (LAS v4i*)l, 0, 0);
#else
    *(v8h*)l = *(const v8h*)g;
#endif
}

__device__ inline void async_wait0() {
#if HAVE_ASYNC_LDS
#if __has_builtin(__builtin_amdgcn_s_wait_asynccnt)
    __builtin_amdgcn_s_wait_asynccnt(0);
#else
    asm volatile("s_wait_asynccnt 0x0" ::: "memory");
#endif
#endif
}

__device__ inline v8f zero8() {
    v8f z;
#pragma unroll
    for (int i = 0; i < 8; ++i) z[i] = 0.0f;
    return z;
}

// 16x32 f16 fragment for one half-wave from a row of 32 halfs (LDS or global).
// CDNA5 16-bit A layout: lanes 0-15 hold K={0..7,16..23}, lanes 16-31 K={8..15,24..31}.
__device__ inline v16h frag32(const _Float16* __restrict__ row, int half) {
    const v8h lo = *(const v8h*)(row + half * 8);
    const v8h hi = *(const v8h*)(row + 16 + half * 8);
    v16h f;
#pragma unroll
    for (int i = 0; i < 8; ++i) { f[i] = lo[i]; f[i + 8] = hi[i]; }
    return f;
}

__device__ inline v8f wmma_f16(v16h a, v16h b, v8f c) {
    return __builtin_amdgcn_wmma_f32_16x16x32_f16(false, a, false, b, (short)0, c,
                                                  false, false);
}

// ---------------------------------------------------------------------------
// C[M,N] = A[M,K] * B[N,K]^T.  B always fp32 (weights). A fp32 or f16 (AF16).
// EPI: 0 = fp32 row-major -> Cf
//      1 = f16  row-major -> Ch
//      2 = KV scatter: fp32 -> Cf at [B,H,S,DK] (s=Sc+t), f16 -> Ch same layout
//      3 = KV scatter: fp32 -> Cf at [B,H,S,DK],          f16 -> Ch [B,H,DK,S]
// Block 256 thr (8 waves), tile 64x128x32, wave 32x32, LDS double-buffered.
// ---------------------------------------------------------------------------
#define GM 64
#define GN 128
#define GK 32

template <bool AF16, int EPI>
__global__ __launch_bounds__(256) void gemm_kernel(
    const void* __restrict__ Ap, const float* __restrict__ Bw,
    float* __restrict__ Cf, _Float16* __restrict__ Ch,
    int M, int N, int K, int T, int H, int S, int Sc, int DKd) {
    __shared__ _Float16 As[2][GM][GK];
    __shared__ _Float16 Bs[2][GN][GK];

    const float*    A32 = (const float*)Ap;
    const _Float16* A16 = (const _Float16*)Ap;

    const int tid  = threadIdx.x;
    const int wave = tid >> 5;
    const int lane = tid & 31;
    const int half = lane >> 4;
    const int lrow = lane & 15;
    const int m0 = blockIdx.y * GM;
    const int n0 = blockIdx.x * GN;
    const int wm = (wave >> 2) * 32;
    const int wn = (wave & 3) * 32;

    v8f acc[2][2];
#pragma unroll
    for (int r = 0; r < 2; ++r)
#pragma unroll
        for (int c = 0; c < 2; ++c) acc[r][c] = zero8();

    // ---- prologue: stage tile k0=0 into buffer 0
    {
        if constexpr (AF16) {
            const int row = tid >> 2, c8 = (tid & 3) << 3;
            *(v8h*)&As[0][row][c8] = *(const v8h*)(A16 + (size_t)(m0 + row) * K + c8);
        } else {
#pragma unroll
            for (int v = 0; v < 2; ++v) {
                const int e4 = tid + v * 256, row = e4 >> 3, c4 = (e4 & 7) << 2;
                const float4 a = *(const float4*)(A32 + (size_t)(m0 + row) * K + c4);
                As[0][row][c4 + 0] = (_Float16)a.x; As[0][row][c4 + 1] = (_Float16)a.y;
                As[0][row][c4 + 2] = (_Float16)a.z; As[0][row][c4 + 3] = (_Float16)a.w;
            }
        }
#pragma unroll
        for (int v = 0; v < 4; ++v) {
            const int e4 = tid + v * 256, row = e4 >> 3, c4 = (e4 & 7) << 2;
            const float4 b = *(const float4*)(Bw + (size_t)(n0 + row) * K + c4);
            Bs[0][row][c4 + 0] = (_Float16)b.x; Bs[0][row][c4 + 1] = (_Float16)b.y;
            Bs[0][row][c4 + 2] = (_Float16)b.z; Bs[0][row][c4 + 3] = (_Float16)b.w;
        }
    }
    __syncthreads();

    int buf = 0;
    for (int k0 = 0; k0 < K; k0 += GK) {
        const bool hasNext = (k0 + GK) < K;
        const int kn = k0 + GK;

        // ---- issue next-tile global loads BEFORE the math (latency hiding)
        float4 arf[2]; v8h arh; float4 brf[4];
        if (hasNext) {
            if constexpr (AF16) {
                arh = *(const v8h*)(A16 + (size_t)(m0 + (tid >> 2)) * K + kn +
                                    ((tid & 3) << 3));
            } else {
#pragma unroll
                for (int v = 0; v < 2; ++v) {
                    const int e4 = tid + v * 256;
                    arf[v] = *(const float4*)(A32 + (size_t)(m0 + (e4 >> 3)) * K + kn +
                                              ((e4 & 7) << 2));
                }
            }
#pragma unroll
            for (int v = 0; v < 4; ++v) {
                const int e4 = tid + v * 256;
                brf[v] = *(const float4*)(Bw + (size_t)(n0 + (e4 >> 3)) * K + kn +
                                          ((e4 & 7) << 2));
            }
            if (kn + GK < K)
                __builtin_prefetch(Bw + (size_t)(n0 + (tid >> 3)) * K + kn + GK, 0, 3);
        }

        // ---- compute on current buffer
        v16h af[2], bf[2];
#pragma unroll
        for (int r = 0; r < 2; ++r) af[r] = frag32(&As[buf][wm + r * 16 + lrow][0], half);
#pragma unroll
        for (int c = 0; c < 2; ++c) bf[c] = frag32(&Bs[buf][wn + c * 16 + lrow][0], half);
#pragma unroll
        for (int r = 0; r < 2; ++r)
#pragma unroll
            for (int c = 0; c < 2; ++c) acc[r][c] = wmma_f16(af[r], bf[c], acc[r][c]);

        // ---- convert & park next tile into the other buffer
        if (hasNext) {
            const int nb = buf ^ 1;
            if constexpr (AF16) {
                *(v8h*)&As[nb][tid >> 2][(tid & 3) << 3] = arh;
            } else {
#pragma unroll
                for (int v = 0; v < 2; ++v) {
                    const int e4 = tid + v * 256, row = e4 >> 3, c4 = (e4 & 7) << 2;
                    As[nb][row][c4 + 0] = (_Float16)arf[v].x;
                    As[nb][row][c4 + 1] = (_Float16)arf[v].y;
                    As[nb][row][c4 + 2] = (_Float16)arf[v].z;
                    As[nb][row][c4 + 3] = (_Float16)arf[v].w;
                }
            }
#pragma unroll
            for (int v = 0; v < 4; ++v) {
                const int e4 = tid + v * 256, row = e4 >> 3, c4 = (e4 & 7) << 2;
                Bs[nb][row][c4 + 0] = (_Float16)brf[v].x;
                Bs[nb][row][c4 + 1] = (_Float16)brf[v].y;
                Bs[nb][row][c4 + 2] = (_Float16)brf[v].z;
                Bs[nb][row][c4 + 3] = (_Float16)brf[v].w;
            }
        }
        __syncthreads();
        buf ^= 1;
    }

    // ---- epilogue
#pragma unroll
    for (int r = 0; r < 2; ++r)
#pragma unroll
        for (int c = 0; c < 2; ++c)
#pragma unroll
            for (int i = 0; i < 8; ++i) {
                const int m = m0 + wm + r * 16 + i + half * 8;
                const int n = n0 + wn + c * 16 + lrow;
                const float val = acc[r][c][i];
                if constexpr (EPI == 0) {
                    Cf[(size_t)m * N + n] = val;
                } else if constexpr (EPI == 1) {
                    Ch[(size_t)m * N + n] = (_Float16)val;
                } else {
                    const int b = m / T, t = m % T;
                    const int h = n / DKd, d = n % DKd;
                    const size_t bh = (size_t)b * H + h;
                    const size_t s = (size_t)Sc + t;
                    Cf[(bh * S + s) * DKd + d] = val;
                    if constexpr (EPI == 2)
                        Ch[(bh * S + s) * DKd + d] = (_Float16)val;
                    else
                        Ch[(bh * DKd + d) * S + s] = (_Float16)val;
                }
            }
}

// ---------------------------------------------------------------------------
// Cache half of the concat (s < Sc): fp32 -> d_out and f16 -> attention copy.
// VT: f16 copy is [B,H,DK,S] (transposed) for V, [B,H,S,DK] for K.
// ---------------------------------------------------------------------------
template <bool VT>
__global__ void concat_cache_kernel(const float* __restrict__ cache,
                                    float* __restrict__ outf,
                                    _Float16* __restrict__ outh, int Bn, int H,
                                    int Sc, int S, int DKd) {
    const size_t total4 = (size_t)Bn * H * Sc * DKd / 4;
    const size_t e4 = (size_t)blockIdx.x * blockDim.x + threadIdx.x;
    if (e4 >= total4) return;
    const size_t e = e4 * 4;
    const int d = (int)(e % DKd);
    const size_t r = e / DKd;
    const int s = (int)(r % Sc);
    const size_t bh = r / Sc;
    const float4 val = *(const float4*)(cache + e);
    const size_t o = (bh * S + s) * DKd + d;
    *(float4*)(outf + o) = val;
    if constexpr (!VT) {
        v4h hv;
        hv[0] = (_Float16)val.x; hv[1] = (_Float16)val.y;
        hv[2] = (_Float16)val.z; hv[3] = (_Float16)val.w;
        *(v4h*)(outh + o) = hv;
    } else {
        outh[(bh * DKd + d + 0) * S + s] = (_Float16)val.x;
        outh[(bh * DKd + d + 1) * S + s] = (_Float16)val.y;
        outh[(bh * DKd + d + 2) * S + s] = (_Float16)val.z;
        outh[(bh * DKd + d + 3) * S + s] = (_Float16)val.w;
    }
}

// ---------------------------------------------------------------------------
// Causal flash attention, all-f16 inputs. Block = (b,h) x 128 query rows,
// 8 waves x 16 rows, S-tiles of 32, DK=128. K/V tiles staged with
// global_load_async_to_lds_b128 when available.
// ---------------------------------------------------------------------------
#define ADK 128
#define SQT 128
#define STL 32

__global__ __launch_bounds__(256) void attn_kernel(
    const _Float16* __restrict__ Qh,  // [B*T, C]
    const _Float16* __restrict__ Kh,  // [B,H,S,DK]
    const _Float16* __restrict__ Vh,  // [B,H,DK,S]  (pre-transposed)
    _Float16* __restrict__ AOh,       // [B*T, C]
    int T, int H, int S) {
    const int bh = blockIdx.y;
    const int b = bh / H, h = bh % H;
    const int q0 = blockIdx.x * SQT;
    const int tid  = threadIdx.x;
    const int wave = tid >> 5;
    const int lane = tid & 31;
    const int half = lane >> 4;
    const int lrow = lane & 15;
    const int C = H * ADK;

    __shared__ _Float16 Kt[STL][ADK];    // s-major K tile
    __shared__ _Float16 Vt[ADK][STL];    // d-major V tile
    __shared__ _Float16 Pt[8][16][STL];  // per-wave P bounce (C->A layout)

    // Q fragments (16 rows x 128) from global f16.
    const _Float16* Qbase =
        Qh + (size_t)(b * T + q0 + wave * 16 + lrow) * C + h * ADK;
    v16h qf[4];
#pragma unroll
    for (int kk = 0; kk < 4; ++kk) qf[kk] = frag32(Qbase + kk * 32, half);

    v8f o[8];
#pragma unroll
    for (int i = 0; i < 8; ++i) o[i] = zero8();
    float mi[8], li[8];
#pragma unroll
    for (int i = 0; i < 8; ++i) { mi[i] = -1e30f; li[i] = 0.0f; }

    const float scl = 0.088388347648318447f;  // 1/sqrt(128)
    const _Float16* Kbh = Kh + (size_t)bh * S * ADK;
    const _Float16* Vbh = Vh + (size_t)bh * ADK * S;
    const int nIt = (q0 + SQT) / STL;  // causal bound (uniform per block)

    for (int it = 0; it < nIt; ++it) {
        const int s0 = it * STL;
        // K tile: 32x128 f16 contiguous in Kh -> flat 512 x 16B chunks.
        _Float16* KtF = &Kt[0][0];
        const _Float16* Ksrc = Kbh + (size_t)s0 * ADK;
#pragma unroll
        for (int v = 0; v < 2; ++v) {
            const int cc = tid + v * 256;
            async_cp16(Ksrc + cc * 8, KtF + cc * 8);
        }
        // V tile: row d of Vt = Vh[bh][d][s0..s0+31] (contiguous 64B).
#pragma unroll
        for (int v = 0; v < 2; ++v) {
            const int cc = tid + v * 256;
            const int d = cc >> 2, c8 = (cc & 3) << 3;
            async_cp16(Vbh + (size_t)d * S + s0 + c8, &Vt[d][c8]);
        }
        async_wait0();
        __syncthreads();

        // Scores: two 16x16 tiles over s, K-dim = 128.
        float pbuf[2][8];
        float tmax[8];
#pragma unroll
        for (int i = 0; i < 8; ++i) tmax[i] = -1e30f;
#pragma unroll
        for (int j = 0; j < 2; ++j) {
            v8f acc = zero8();
#pragma unroll
            for (int kk = 0; kk < 4; ++kk) {
                const v16h bf = frag32(&Kt[j * 16 + lrow][kk * 32], half);
                acc = wmma_f16(qf[kk], bf, acc);
            }
            const int sg = s0 + j * 16 + lrow;
#pragma unroll
            for (int i = 0; i < 8; ++i) {
                const int tg = q0 + wave * 16 + i + half * 8;
                float sc = acc[i] * scl;
                sc = (sg > tg) ? -1e30f : sc;
                pbuf[j][i] = sc;
                tmax[i] = fmaxf(tmax[i], sc);
            }
        }

        // Online softmax (rows live across 16 lanes of a half-wave).
#pragma unroll
        for (int i = 0; i < 8; ++i) {
            float v = tmax[i];
            v = fmaxf(v, __shfl_xor(v, 1, 16));
            v = fmaxf(v, __shfl_xor(v, 2, 16));
            v = fmaxf(v, __shfl_xor(v, 4, 16));
            v = fmaxf(v, __shfl_xor(v, 8, 16));
            const float mnew = fmaxf(mi[i], v);
            const float fac = __expf(mi[i] - mnew);
            float rsum = 0.0f;
#pragma unroll
            for (int j = 0; j < 2; ++j) {
                const float p = __expf(pbuf[j][i] - mnew);
                rsum += p;
                Pt[wave][i + half * 8][j * 16 + lrow] = (_Float16)p;
            }
            rsum += __shfl_xor(rsum, 1, 16);
            rsum += __shfl_xor(rsum, 2, 16);
            rsum += __shfl_xor(rsum, 4, 16);
            rsum += __shfl_xor(rsum, 8, 16);
            li[i] = li[i] * fac + rsum;
            mi[i] = mnew;
#pragma unroll
            for (int jn = 0; jn < 8; ++jn) o[jn][i] *= fac;
        }

        // O += P(16x32) @ V(32x128)
        const v16h pf = frag32(&Pt[wave][lrow][0], half);
#pragma unroll
        for (int jn = 0; jn < 8; ++jn) {
            const v16h bf = frag32(&Vt[jn * 16 + lrow][0], half);
            o[jn] = wmma_f16(pf, bf, o[jn]);
        }
        __syncthreads();
    }

    // Normalize, write f16 attention output for the Wo GEMM.
    _Float16* AObase = AOh + (size_t)(b * T) * C + h * ADK;
#pragma unroll
    for (int i = 0; i < 8; ++i) {
        const float inv = (li[i] > 0.0f) ? 1.0f / li[i] : 0.0f;
        const int t = q0 + wave * 16 + i + half * 8;
#pragma unroll
        for (int jn = 0; jn < 8; ++jn)
            AObase[(size_t)t * C + jn * 16 + lrow] = (_Float16)(o[jn][i] * inv);
    }
}

// ---------------------------------------------------------------------------
extern "C" void kernel_launch(void* const* d_in, const int* in_sizes, int n_in,
                              void* d_out, int out_size, void* d_ws, size_t ws_size,
                              hipStream_t stream) {
    const float* x       = (const float*)d_in[0];
    const float* cache_k = (const float*)d_in[1];
    const float* cache_v = (const float*)d_in[2];
    const float* Wq      = (const float*)d_in[3];
    const float* Wk      = (const float*)d_in[4];
    const float* Wv      = (const float*)d_in[5];
    const float* Wo      = (const float*)d_in[6];

    const int Bn = 2, T = 1024, C = 2048, H = 16, Sc = 1024, DKc = 128;
    const int S = Sc + T;
    const int M = Bn * T;
    const size_t KV = (size_t)Bn * H * S * DKc;

    // f16 workspace: Qh | Kh | Vh(T) | AOh  = 8+16+16+8 = 48 MB
    _Float16* Qh  = (_Float16*)d_ws;
    _Float16* Kh  = Qh + (size_t)M * C;
    _Float16* Vh  = Kh + KV;
    _Float16* AOh = Vh + KV;

    float* Out  = (float*)d_out;
    float* Kout = Out + (size_t)M * C;
    float* Vout = Kout + KV;

    dim3 gblk(C / GN, M / GM);

    // QKV projections; K/V scatter straight into d_out concat positions.
    gemm_kernel<false, 1><<<gblk, 256, 0, stream>>>(x, Wq, nullptr, Qh, M, C, C,
                                                    T, H, S, Sc, DKc);
    gemm_kernel<false, 2><<<gblk, 256, 0, stream>>>(x, Wk, Kout, Kh, M, C, C,
                                                    T, H, S, Sc, DKc);
    gemm_kernel<false, 3><<<gblk, 256, 0, stream>>>(x, Wv, Vout, Vh, M, C, C,
                                                    T, H, S, Sc, DKc);

    // Cache half of the concat (fp32 out + f16 attention copies).
    const size_t total4 = (size_t)Bn * H * Sc * DKc / 4;
    const int cblocks = (int)((total4 + 255) / 256);
    concat_cache_kernel<false><<<cblocks, 256, 0, stream>>>(cache_k, Kout, Kh,
                                                            Bn, H, Sc, S, DKc);
    concat_cache_kernel<true><<<cblocks, 256, 0, stream>>>(cache_v, Vout, Vh,
                                                           Bn, H, Sc, S, DKc);

    dim3 ablk(T / SQT, Bn * H);
    attn_kernel<<<ablk, 256, 0, stream>>>(Qh, Kh, Vh, AOh, T, H, S);

    gemm_kernel<true, 0><<<gblk, 256, 0, stream>>>(AOh, Wo, Out, nullptr, M, C, C,
                                                   T, H, S, Sc, DKc);
}
